// RNNCell_Encoder_3951369912686
// MI455X (gfx1250) — matrix operand order repaired and checked by reading, not compiled, not run on hardware
//
#include <hip/hip_runtime.h>
#include <hip/hip_bf16.h>

// ---------------------------------------------------------------------------
// Tanh RNN scan for MI455X (gfx1250, wave32, WMMA).
//   h_{t+1} = tanh( X_t @ W_ih^T + h_t @ W_hh^T + (b_ih + b_hh) )
//
//   1) pack_wmma_b : W_ih/W_hh -> exact 32x16 f16 B-operand lane layout.
//   2) gemm_xproj  : all 512 timesteps' input projections + bias in ONE wide
//      WMMA GEMM. A k-tiles pinned in 80 VGPRs (X read exactly once, 315MB
//      ~ 13.5us @ 23.3TB/s). B slabs (10KB/nt) staged in LDS, double-
//      buffered, via CDNA5 async global->LDS loads (ASYNCcnt).
//   3) persistent scan: ONE launch, 608 single-wave blocks, 512 steps with a
//      sense-reversing device-scope atomic grid barrier. Each block async-
//      stages its W_hh slab into LDS ONCE, so per-step serial work is
//      10 WMMAs fed from LDS + the unavoidable cross-WGP h reads.
// ---------------------------------------------------------------------------

typedef __attribute__((ext_vector_type(16))) _Float16 v16h;
typedef __attribute__((ext_vector_type(8)))  _Float16 v8h;
typedef __attribute__((ext_vector_type(8)))  float    v8f;
typedef __attribute__((ext_vector_type(4)))  float    v4f;

#define HID   300
#define BATCH 512
#define SEQ   512
#define KPAD  320           // 10 k-tiles of 32
#define NT    19            // n-tiles (ceil 300/16)
#define MT    32            // m-tiles (512/16)
#define KT    10
#define NBLK  (NT * MT)     // 608 persistent single-wave blocks
#define GMT   (SEQ * BATCH / 16)   // 16384 global m-tiles for xproj

// ---- workspace layout (bytes) ----
#define OFF_WIH   0u
#define OFF_WHH   194560u
#define OFF_BIAS  389120u
#define OFF_SYNC  390336u
#define OFF_H0    390400u
#define OFF_H1    718080u
#define OFF_XP    1045760u
#define XP_BYTES  ((size_t)SEQ * NBLK * 512)          // 159,383,552
#define WS_NEED_XP (OFF_XP + XP_BYTES)

__device__ __forceinline__ v8f wmma16(v16h a, v16h b, v8f c) {
    return __builtin_amdgcn_wmma_f32_16x16x32_f16(false, a, false, b,
                                                  (short)0, c, false, false);
}

// Issue one async 16B global->LDS copy (per active lane). ASYNCcnt-tracked.
__device__ __forceinline__ void async_copy16(const _Float16* gsrc, _Float16* ldst) {
    unsigned long long g = (unsigned long long)(uintptr_t)gsrc;
    unsigned int       l = (unsigned int)(uintptr_t)ldst;  // flat low 32 = LDS off
    asm volatile("global_load_async_to_lds_b128 %0, %1, off"
                 :: "v"(l), "v"(g) : "memory");
}
__device__ __forceinline__ void wait_async0() {
    asm volatile("s_wait_asynccnt 0x0" ::: "memory");
}

// A-operand 16x32 f16 layout: lanes 0-15 carry row M=lane, K runs [+0..7],
// [+16..23]; lanes 16-31 carry row M=lane-16, K runs [+8..15], [+24..31].
__device__ __forceinline__ v16h load_a_from_f32(const float* __restrict__ xrow,
                                                int kt, int half16) {
    const int kbase = kt * 32 + half16 * 8;
    v16h ax;
    if (kt < KT - 1) {
        const v4f x0 = *(const v4f*)(xrow + kbase);
        const v4f x1 = *(const v4f*)(xrow + kbase + 4);
        const v4f x2 = *(const v4f*)(xrow + kbase + 16);
        const v4f x3 = *(const v4f*)(xrow + kbase + 20);
#pragma unroll
        for (int c = 0; c < 4; ++c) {
            ax[c]      = (_Float16)x0[c];
            ax[4 + c]  = (_Float16)x1[c];
            ax[8 + c]  = (_Float16)x2[c];
            ax[12 + c] = (_Float16)x3[c];
        }
    } else {
#pragma unroll
        for (int j = 0; j < 8; ++j) {
            int k1 = kbase + j;
            int k2 = kbase + 16 + j;
            ax[j]     = (k1 < HID) ? (_Float16)xrow[k1] : (_Float16)0.0f;
            ax[8 + j] = (k2 < HID) ? (_Float16)xrow[k2] : (_Float16)0.0f;
        }
    }
    return ax;
}

__device__ __forceinline__ v16h load_a_from_h(const _Float16* __restrict__ hrow,
                                              int kt, int half16) {
    const int kbase = kt * 32 + half16 * 8;
    union { v16h v; v8h h[2]; } ah;
    ah.h[0] = *(const v8h*)(hrow + kbase);
    ah.h[1] = *(const v8h*)(hrow + kbase + 16);
    return ah.v;
}

// Read one B tile (32x16 f16) from an LDS slab: two ds_load_b128 per lane.
__device__ __forceinline__ v16h load_b_lds(const _Float16* slab, int kt, int lane) {
    union { v16h v; v8h h[2]; } b;
    b.h[0] = *(const v8h*)(slab + kt * 512 + lane * 16);
    b.h[1] = *(const v8h*)(slab + kt * 512 + lane * 16 + 8);
    return b.v;
}

// Sense-reversing grid barrier (device scope). Single-wave blocks: lane 0
// arrives/polls, wave reconvergence holds the other lanes.
__device__ __forceinline__ void grid_barrier(unsigned* __restrict__ sync) {
    __threadfence();
    if (threadIdx.x == 0) {
        unsigned gen = __hip_atomic_load(&sync[1], __ATOMIC_RELAXED,
                                         __HIP_MEMORY_SCOPE_AGENT);
        unsigned prev = __hip_atomic_fetch_add(&sync[0], 1u, __ATOMIC_ACQ_REL,
                                               __HIP_MEMORY_SCOPE_AGENT);
        if (prev == (unsigned)(NBLK - 1)) {
            __hip_atomic_store(&sync[0], 0u, __ATOMIC_RELAXED,
                               __HIP_MEMORY_SCOPE_AGENT);
            __hip_atomic_fetch_add(&sync[1], 1u, __ATOMIC_RELEASE,
                                   __HIP_MEMORY_SCOPE_AGENT);
        } else {
            while (__hip_atomic_load(&sync[1], __ATOMIC_ACQUIRE,
                                     __HIP_MEMORY_SCOPE_AGENT) == gen) {
                __builtin_amdgcn_s_sleep(2);
            }
        }
    }
    __threadfence();
}

// ---------------------------------------------------------------------------
// W [300x300] -> B-operand tiles: dst[(kt*NT+nt)*512 + lane*16 + j] = W[n][k],
// n = nt*16 + (lane&15), k = kt*32 + (lane>>4)*8 + (j&7) + (j>>3)*16, zero padded.
// ---------------------------------------------------------------------------
__global__ void pack_wmma_b(const float* __restrict__ W, _Float16* __restrict__ dst) {
    int idx = blockIdx.x * blockDim.x + threadIdx.x;
    if (idx >= KT * NT * 512) return;
    int j    = idx & 15;
    int lane = (idx >> 4) & 31;
    int tile = idx >> 9;
    int nt   = tile % NT;
    int kt   = tile / NT;
    int n = nt * 16 + (lane & 15);
    int k = kt * 32 + (lane >> 4) * 8 + (j & 7) + (j >> 3) * 16;
    float v = (n < HID && k < HID) ? W[n * HID + k] : 0.0f;
    dst[idx] = (_Float16)v;
}

__global__ void init_state(const float* __restrict__ b_ih,
                           const float* __restrict__ b_hh,
                           float* __restrict__ bias,
                           _Float16* __restrict__ h0,
                           _Float16* __restrict__ h1,
                           unsigned* __restrict__ sync) {
    int idx = blockIdx.x * blockDim.x + threadIdx.x;
    if (idx < HID) bias[idx] = b_ih[idx] + b_hh[idx];
    if (idx < 2) sync[idx] = 0u;
    if (idx < BATCH * KPAD) {
        h0[idx] = (_Float16)0.0f;
        h1[idx] = (_Float16)0.0f;
    }
}

// ---------------------------------------------------------------------------
// xp[(gmt*NT + nt)*256 + lane*8 + r] = f16( X@W_ih^T + bias ) in C-tile order.
// 8 waves / block; A k-tiles pinned in registers; B slab (10KB) double-
// buffered in LDS via async global->LDS loads, prefetching nt+1.
// ---------------------------------------------------------------------------
__global__ __launch_bounds__(256) void gemm_xproj(
    const float*    __restrict__ x,     // [SEQ*BATCH][300]
    const _Float16* __restrict__ wih,
    const float*    __restrict__ bias,
    _Float16*       __restrict__ xp) {
    __shared__ _Float16 sB[2][KT * 512];          // 2 x 10KB B slabs

    const int tid    = threadIdx.x;
    const int wave   = tid >> 5;
    const int lane   = tid & 31;
    const int gmt    = blockIdx.x * 8 + wave;
    const int half16 = lane >> 4;
    const int l16    = lane & 15;
    const float* xrow = x + (size_t)(gmt * 16 + l16) * HID;

    v16h ax[KT];
#pragma unroll
    for (int kt = 0; kt < KT; ++kt) ax[kt] = load_a_from_f32(xrow, kt, half16);

    auto stage = [&](int buf, int nt) {
        for (int c = tid; c < KT * 64; c += 256) {
            const int kt = c >> 6, inner = c & 63;
            async_copy16(wih + (size_t)(kt * NT + nt) * 512 + inner * 8,
                         &sB[buf][kt * 512 + inner * 8]);
        }
    };

    stage(0, 0);
    wait_async0();
    __syncthreads();

#pragma unroll 1
    for (int nt = 0; nt < NT; ++nt) {
        const int buf = nt & 1;
        if (nt + 1 < NT) stage(buf ^ 1, nt + 1);

        v8f acc = {};
#pragma unroll
        for (int kt = 0; kt < KT; ++kt)
            acc = wmma16(ax[kt], load_b_lds(sB[buf], kt, lane), acc);

        const int n = nt * 16 + l16;
        const float bv = (n < HID) ? bias[n] : 0.0f;
        v8h st;
#pragma unroll
        for (int r = 0; r < 8; ++r) st[r] = (_Float16)(acc[r] + bv);
        *(v8h*)(xp + ((size_t)gmt * NT + nt) * 256 + lane * 8) = st;

        wait_async0();      // prefetched slab landed
        __syncthreads();    // everyone done reading current slab
    }
}

// ---------------------------------------------------------------------------
// Persistent scan, xp path. Grid (NT, MT), 32 threads, one launch, 512 steps.
// W_hh slab (10KB, fixed nt) async-staged to LDS once; per step:
// acc <- xp tile; 10 LDS-fed WMMAs vs h; tanh; grid barrier.
// ---------------------------------------------------------------------------
__global__ __launch_bounds__(32) void scan_xp(
    const _Float16* __restrict__ xp,
    _Float16* h0, _Float16* h1,
    const _Float16* __restrict__ whh,
    float* __restrict__ out,
    unsigned* __restrict__ sync) {
    __shared__ _Float16 sW[KT * 512];             // this block's W_hh slab
    const int nt = blockIdx.x, mt = blockIdx.y;
    const int lane = threadIdx.x;
    const int half16 = lane >> 4, l16 = lane & 15;
    const int n = nt * 16 + l16;
    const int mrow = mt * 16 + l16;
    _Float16* hc = h0;
    _Float16* hn = h1;

    for (int c = lane; c < KT * 64; c += 32) {     // uniform: 20 issues/lane
        const int kt = c >> 6, inner = c & 63;
        async_copy16(whh + (size_t)(kt * NT + nt) * 512 + inner * 8,
                     &sW[kt * 512 + inner * 8]);
    }
    wait_async0();   // single wave: no s_barrier needed for LDS visibility

    for (int t = 0; t < SEQ; ++t) {
        const v8h p = *(const v8h*)(xp + (((size_t)(t * MT + mt)) * NT + nt) * 256 + lane * 8);
        v8f acc;
#pragma unroll
        for (int r = 0; r < 8; ++r) acc[r] = (float)p[r];

        const _Float16* hrow = hc + (size_t)mrow * KPAD;
#pragma unroll
        for (int kt = 0; kt < KT; ++kt)
            acc = wmma16(load_a_from_h(hrow, kt, half16),
                         load_b_lds(sW, kt, lane), acc);

#pragma unroll
        for (int r = 0; r < 8; ++r) {
            float v = tanhf(acc[r]);
            int m = mt * 16 + r + 8 * half16;
            if (n < HID) {
                hn[(size_t)m * KPAD + n] = (_Float16)v;
                if (t == SEQ - 1) out[(size_t)m * HID + n] = v;
            }
        }
        grid_barrier(sync);
        _Float16* tmp = hc; hc = hn; hn = tmp;
    }
}

// Persistent scan, fused fallback (no xp buffer): both slabs in LDS (20KB),
// 20 WMMAs/step.
__global__ __launch_bounds__(32) void scan_fused(
    const float*    __restrict__ x,     // [SEQ][BATCH][300]
    _Float16* h0, _Float16* h1,
    const _Float16* __restrict__ wih,
    const _Float16* __restrict__ whh,
    const float*    __restrict__ bias,
    float* __restrict__ out,
    unsigned* __restrict__ sync) {
    __shared__ _Float16 sW[2][KT * 512];          // [0]=W_ih, [1]=W_hh slabs
    const int nt = blockIdx.x, mt = blockIdx.y;
    const int lane = threadIdx.x;
    const int half16 = lane >> 4, l16 = lane & 15;
    const int n = nt * 16 + l16;
    const int mrow = mt * 16 + l16;
    const float bv = (n < HID) ? bias[n] : 0.0f;
    _Float16* hc = h0;
    _Float16* hn = h1;

    for (int c = lane; c < KT * 64; c += 32) {
        const int kt = c >> 6, inner = c & 63;
        async_copy16(wih + (size_t)(kt * NT + nt) * 512 + inner * 8,
                     &sW[0][kt * 512 + inner * 8]);
        async_copy16(whh + (size_t)(kt * NT + nt) * 512 + inner * 8,
                     &sW[1][kt * 512 + inner * 8]);
    }
    wait_async0();

    for (int t = 0; t < SEQ; ++t) {
        v8f acc;
#pragma unroll
        for (int r = 0; r < 8; ++r) acc[r] = bv;

        const float* xrow = x + ((size_t)t * BATCH + mrow) * HID;
        const _Float16* hrow = hc + (size_t)mrow * KPAD;
#pragma unroll
        for (int kt = 0; kt < KT; ++kt) {
            acc = wmma16(load_a_from_f32(xrow, kt, half16),
                         load_b_lds(sW[0], kt, lane), acc);
            acc = wmma16(load_a_from_h(hrow, kt, half16),
                         load_b_lds(sW[1], kt, lane), acc);
        }
#pragma unroll
        for (int r = 0; r < 8; ++r) {
            float v = tanhf(acc[r]);
            int m = mt * 16 + r + 8 * half16;
            if (n < HID) {
                hn[(size_t)m * KPAD + n] = (_Float16)v;
                if (t == SEQ - 1) out[(size_t)m * HID + n] = v;
            }
        }
        grid_barrier(sync);
        _Float16* tmp = hc; hc = hn; hn = tmp;
    }
}

extern "C" void kernel_launch(void* const* d_in, const int* in_sizes, int n_in,
                              void* d_out, int out_size, void* d_ws, size_t ws_size,
                              hipStream_t stream) {
    const float* inputs = (const float*)d_in[0];
    const float* W_ih   = (const float*)d_in[1];
    const float* W_hh   = (const float*)d_in[2];
    const float* b_ih   = (const float*)d_in[3];
    const float* b_hh   = (const float*)d_in[4];

    char* ws = (char*)d_ws;
    _Float16* wih_p = (_Float16*)(ws + OFF_WIH);
    _Float16* whh_p = (_Float16*)(ws + OFF_WHH);
    float*    bias  = (float*)   (ws + OFF_BIAS);
    unsigned* sync  = (unsigned*)(ws + OFF_SYNC);
    _Float16* h0    = (_Float16*)(ws + OFF_H0);
    _Float16* h1    = (_Float16*)(ws + OFF_H1);
    _Float16* xp    = (_Float16*)(ws + OFF_XP);

    const int packN = KT * NT * 512;
    pack_wmma_b<<<(packN + 255) / 256, 256, 0, stream>>>(W_ih, wih_p);
    pack_wmma_b<<<(packN + 255) / 256, 256, 0, stream>>>(W_hh, whh_p);
    init_state<<<(BATCH * KPAD + 255) / 256, 256, 0, stream>>>(b_ih, b_hh, bias, h0, h1, sync);

    if (ws_size >= WS_NEED_XP) {
        gemm_xproj<<<GMT / 8, 256, 0, stream>>>(inputs, wih_p, bias, xp);
        scan_xp<<<dim3(NT, MT), 32, 0, stream>>>(xp, h0, h1, whh_p, (float*)d_out, sync);
    } else {
        scan_fused<<<dim3(NT, MT), 32, 0, stream>>>(inputs, h0, h1, wih_p, whh_p,
                                                    bias, (float*)d_out, sync);
    }
}
